// MRNNFixD_57887569216035
// MI455X (gfx1250) — matrix operand order repaired and checked by reading, not compiled, not run on hardware
//
#include <hip/hip_runtime.h>

// ---------------------------------------------------------------------------
// MRNNFixD on MI455X (gfx1250, wave32, WMMA).
//   T=1024 B=128 IN=32 H=512 OUT=32 K=64
// Pipeline per layer: frac-filter -> bf16 WMMA GEMM (pre) -> persistent
// recurrence kernel (WMMA per step, grid phase-barrier + cluster barrier)
// -> bf16 WMMA readout GEMM. Final step readout is a tiny scalar kernel.
// ---------------------------------------------------------------------------

static constexpr int Tn   = 1024;
static constexpr int Bn   = 128;
static constexpr int INn  = 32;
static constexpr int Hn   = 512;
static constexpr int OUTn = 32;
static constexpr int Kn   = 64;
static constexpr int BHn  = Bn * Hn;           // 65536
static constexpr int TBn  = Tn * Bn;           // 131072

typedef __attribute__((ext_vector_type(16))) __bf16          v16bf;
typedef __attribute__((ext_vector_type(8)))  float           v8f;
typedef __attribute__((ext_vector_type(8)))  unsigned short  ushort8;

union Frag { v16bf bf; ushort8 us[2]; };

__device__ __forceinline__ unsigned short f2bf(float x) {
    union { float f; unsigned u; } v; v.f = x;
    unsigned r = v.u + 0x7FFFu + ((v.u >> 16) & 1u);   // RNE
    return (unsigned short)(r >> 16);
}
__device__ __forceinline__ float bf2f(unsigned short h) {
    union { unsigned u; float f; } v; v.u = ((unsigned)h) << 16;
    return v.f;
}

// ---------------------------------------------------------------------------
// pi coefficients: kern[lag][h], lag 0..K ; kern[0]=1, kern[j]=cumprod((j-1-d)/j)
// ---------------------------------------------------------------------------
__global__ void pi_kernel(const float* __restrict__ b_d, float* __restrict__ kern) {
    int h = threadIdx.x;                       // blockDim.x == Hn
    float d = 0.5f / (1.0f + expf(-b_d[h]));
    kern[h] = 1.0f;
    float p = 1.0f;
    for (int j = 1; j <= Kn; ++j) {
        p *= ((float)j - 1.0f - d) / (float)j;
        kern[(size_t)j * Hn + h] = p;
    }
}

// ---------------------------------------------------------------------------
// Embedding: xe[tb][h] = bf16( inputs[tb,:] . W_emb[h,:] + b_emb[h] )
// ---------------------------------------------------------------------------
__global__ void embed_kernel(const float* __restrict__ in,
                             const float* __restrict__ Wemb,
                             const float* __restrict__ bemb,
                             unsigned short* __restrict__ xe) {
    int tb = blockIdx.x;
    __shared__ float row[INn];
    if (threadIdx.x < INn) row[threadIdx.x] = in[(size_t)tb * INn + threadIdx.x];
    __syncthreads();
    for (int h = threadIdx.x; h < Hn; h += blockDim.x) {
        const float* w = Wemb + (size_t)h * INn;
        float s = bemb[h];
        #pragma unroll
        for (int i = 0; i < INn; ++i) s += row[i] * w[i];
        xe[(size_t)tb * Hn + h] = f2bf(s);
    }
}

// ---------------------------------------------------------------------------
// Causal depthwise filter: xt[t] = sum_{lag<=min(t,K)} kern[lag] * x[t-lag]
// ---------------------------------------------------------------------------
__global__ void filter_kernel(const unsigned short* __restrict__ x,
                              const float* __restrict__ kern,
                              unsigned short* __restrict__ xt) {
    int t   = blockIdx.y;
    int idx = blockIdx.x * blockDim.x + threadIdx.x;  // over B*H
    int h   = idx & (Hn - 1);
    int lmax = (t < Kn) ? t : Kn;
    float s = 0.0f;
    for (int lag = 0; lag <= lmax; ++lag)
        s += kern[(size_t)lag * Hn + h] * bf2f(x[(size_t)(t - lag) * BHn + idx]);
    xt[(size_t)t * BHn + idx] = f2bf(s);
}

// ---------------------------------------------------------------------------
// Bulk GEMM: Out[M,N] = A[M,512](bf16) * W[N,512](f32->bf16)^T + b1 (+ b2)
// Workgroup = 256 thr = 8 wave32. Each wave owns one 16-row M tile and all
// 4 N tiles (BN=64). W stripe is pre-swizzled into LDS in the exact
// v_wmma_f32_16x16x32_bf16 B-fragment layout: lane l holds B[k,n] with
// n = l&15, k = 32*chunk + (l>=16 ? 8 : 0) + {0..7} and +16+{0..7}.
// A fragment layout (from ISA): lane l row m=l&15 holds k = 32*chunk +
// (l>=16 ? 8 : 0) + {0..7} and +16+{0..7}  -> two b128 loads per fragment.
// ---------------------------------------------------------------------------
template <bool BF16OUT>
__global__ __launch_bounds__(256) void gemm_kernel(
        const unsigned short* __restrict__ A,
        const float* __restrict__ W,
        const float* __restrict__ b1,
        const float* __restrict__ b2,
        void* __restrict__ out, int N) {
    const int tid  = threadIdx.x;
    const int lane = tid & 31;
    const int wv   = tid >> 5;
    const int n0   = blockIdx.x * 64;
    const int m0   = blockIdx.y * 128 + wv * 16;

    __shared__ unsigned short bSh[4 * 8192];   // 64 KB, 4 N-tiles of 512x16 bf16
    for (int idx = tid; idx < 4 * 8192; idx += 256) {
        int j = idx >> 13, r = idx & 8191;
        int c = r >> 9, l = (r >> 4) & 31, i = r & 15;
        int k = c * 32 + ((l >= 16) ? 8 : 0) + (i < 8 ? i : 8 + i);
        int n = n0 + j * 16 + (l & 15);
        bSh[j * 8192 + (c * 32 + l) * 16 + i] = f2bf(W[(size_t)n * Hn + k]);
    }
    __syncthreads();

    const int sel  = lane >> 4;
    const int mrow = m0 + (lane & 15);
    v8f acc[4] = {};
    #pragma unroll 4
    for (int kc = 0; kc < 16; ++kc) {
        Frag a;
        const unsigned short* ap = A + (size_t)mrow * Hn + kc * 32 + sel * 8;
        a.us[0] = *(const ushort8*)(ap);
        a.us[1] = *(const ushort8*)(ap + 16);
        #pragma unroll
        for (int j = 0; j < 4; ++j) {
            Frag b;
            const unsigned short* bp = &bSh[j * 8192 + (kc * 32 + lane) * 16];
            b.us[0] = *(const ushort8*)(bp);
            b.us[1] = *(const ushort8*)(bp + 8);
            acc[j] = __builtin_amdgcn_wmma_f32_16x16x32_bf16(
                false, a.bf, false, b.bf, (short)0, acc[j], false, false);
        }
    }

    const int mo = m0 + sel * 8;
    #pragma unroll
    for (int j = 0; j < 4; ++j) {
        int n = n0 + j * 16 + (lane & 15);
        float bias = b1[n] + (b2 ? b2[n] : 0.0f);
        #pragma unroll
        for (int r = 0; r < 8; ++r) {
            float v = acc[j][r] + bias;
            size_t o = (size_t)(mo + r) * N + n;
            if constexpr (BF16OUT) ((unsigned short*)out)[o] = f2bf(v);
            else                   ((float*)out)[o] = v;
        }
    }
}

// ---------------------------------------------------------------------------
// Grid phase barrier for the persistent recurrence kernel.
// Correctness comes from the device-scope atomic counter; the cluster
// barrier (s_barrier_signal/-wait -3) is a fast-path that is an S_NOP when
// the grid is not dispatched as a workgroup cluster (our 16 WGs fit one).
// ---------------------------------------------------------------------------
#define NWG_REC 16
__device__ __forceinline__ void grid_barrier(unsigned* cnt, unsigned target) {
    __threadfence();                         // release my tile stores (dev scope)
    __syncthreads();
    if (threadIdx.x == 0) {
        __hip_atomic_fetch_add(cnt, 1u, __ATOMIC_RELEASE, __HIP_MEMORY_SCOPE_AGENT);
        while (__hip_atomic_load(cnt, __ATOMIC_ACQUIRE, __HIP_MEMORY_SCOPE_AGENT) < target)
            __builtin_amdgcn_s_sleep(8);
    }
    __builtin_amdgcn_s_cluster_barrier();    // NOP outside cluster dispatch
    __syncthreads();
    __threadfence();                         // acquire: drop stale WGP$ lines
}

// ---------------------------------------------------------------------------
// Persistent recurrence: hs[t+1] = tanh(pre[t] + hs[t] * W_hh^T), hs[0]=0.
// 16 WGs x 16 wave32 = 256 waves; wave (g,w) owns tile id g*16+w:
// nt = id>>3 (N tile), mt = id&7 (M tile). Each WG keeps its two 512x16
// W_hh stripes resident in LDS (fragment-swizzled) for all 1024 steps.
// ---------------------------------------------------------------------------
__global__ __launch_bounds__(512, 1) void recur_kernel(
        const float* __restrict__ pre,
        const float* __restrict__ Whh,
        unsigned short* __restrict__ hs,
        unsigned* __restrict__ bar) {
    const int tid  = threadIdx.x;
    const int lane = tid & 31;
    const int wv   = tid >> 5;
    const int g    = blockIdx.x;
    const int tile = g * 16 + wv;
    const int nt   = tile >> 3;
    const int mt   = tile & 7;
    const int m0   = mt * 16, n0 = nt * 16;
    const int ls   = nt & 1;                  // which LDS stripe (nt = 2g + ls)

    __shared__ unsigned short bSh[2 * 8192];  // 32 KB: two 512x16 bf16 stripes
    for (int idx = tid; idx < 2 * 8192; idx += 512) {
        int s = idx >> 13, r = idx & 8191;
        int c = r >> 9, l = (r >> 4) & 31, i = r & 15;
        int k = c * 32 + ((l >= 16) ? 8 : 0) + (i < 8 ? i : 8 + i);
        int n = (2 * g + s) * 16 + (l & 15);
        bSh[s * 8192 + (c * 32 + l) * 16 + i] = f2bf(Whh[(size_t)n * Hn + k]);
    }
    for (int idx = g * 512 + tid; idx < BHn; idx += NWG_REC * 512)
        hs[idx] = 0;                          // h_{-1} = 0
    __syncthreads();

    unsigned phase = 1;
    grid_barrier(bar, NWG_REC * phase); ++phase;

    const int sel  = lane >> 4;
    const int mrow = m0 + (lane & 15);
    const int ncol = n0 + (lane & 15);
    const int mo   = m0 + sel * 8;

    for (int t = 0; t < Tn; ++t) {
        const unsigned short* hp = hs + (size_t)t * BHn;
        const float* pr = pre + ((size_t)t * Bn + mo) * Hn + ncol;
        if (t + 1 < Tn)                       // stream next step's pre slab
            __builtin_prefetch(pre + ((size_t)(t + 1) * Bn + mo) * Hn + ncol, 0, 0);

        v8f acc = {};
        #pragma unroll
        for (int kc = 0; kc < 16; ++kc) {
            Frag a, b;
            const unsigned short* ap = hp + (size_t)mrow * Hn + kc * 32 + sel * 8;
            a.us[0] = *(const ushort8*)(ap);
            a.us[1] = *(const ushort8*)(ap + 16);
            const unsigned short* bp = &bSh[ls * 8192 + (kc * 32 + lane) * 16];
            b.us[0] = *(const ushort8*)(bp);
            b.us[1] = *(const ushort8*)(bp + 8);
            acc = __builtin_amdgcn_wmma_f32_16x16x32_bf16(
                false, a.bf, false, b.bf, (short)0, acc, false, false);
        }
        unsigned short* ho = hs + (size_t)(t + 1) * BHn + (size_t)mo * Hn + ncol;
        #pragma unroll
        for (int r = 0; r < 8; ++r) {
            float v = tanhf(acc[r] + pr[(size_t)r * Hn]);
            ho[(size_t)r * Hn] = f2bf(v);
        }
        grid_barrier(bar, NWG_REC * phase); ++phase;
    }
}

// ---------------------------------------------------------------------------
// Final readout: out[b,o] = hs1[T][b,:] . W_ho1[o,:] + b_ho1[o]
// ---------------------------------------------------------------------------
__global__ void final_kernel(const unsigned short* __restrict__ hT,
                             const float* __restrict__ Who,
                             const float* __restrict__ b,
                             float* __restrict__ out) {
    int o = blockIdx.x * blockDim.x + threadIdx.x;  // B*OUT = 4096
    if (o >= Bn * OUTn) return;
    int bb = o >> 5, oo = o & 31;
    const unsigned short* hr = hT + (size_t)bb * Hn;
    const float* wr = Who + (size_t)oo * Hn;
    float s = b[oo];
    for (int k = 0; k < Hn; ++k) s += bf2f(hr[k]) * wr[k];
    out[o] = s;
}

// ---------------------------------------------------------------------------
// Workspace layout (all offsets multiples of 256B)
// ---------------------------------------------------------------------------
static constexpr size_t SZ_KERN = (size_t)(Kn + 1) * Hn * sizeof(float);   // 133120
static constexpr size_t OFF_BAR = SZ_KERN;
static constexpr size_t OFF_XE  = OFF_BAR + 256;
static constexpr size_t SZ_A16  = (size_t)Tn * Bn * Hn * 2;                // 128 MB
static constexpr size_t OFF_XT  = OFF_XE + SZ_A16;
static constexpr size_t OFF_PRE = OFF_XT + SZ_A16;
static constexpr size_t OFF_HS  = OFF_PRE + (size_t)Tn * Bn * Hn * 4;      // hs: (T+1)*B*H bf16

extern "C" void kernel_launch(void* const* d_in, const int* in_sizes, int n_in,
                              void* d_out, int out_size, void* d_ws, size_t ws_size,
                              hipStream_t stream) {
    (void)in_sizes; (void)n_in; (void)out_size; (void)ws_size;
    const float* inputs = (const float*)d_in[0];
    const float* b_d    = (const float*)d_in[1];
    const float* W_emb  = (const float*)d_in[2];
    const float* b_emb  = (const float*)d_in[3];
    const float* W_ih0  = (const float*)d_in[4];
    const float* b_ih0  = (const float*)d_in[5];
    const float* W_hh0  = (const float*)d_in[6];
    const float* b_hh0  = (const float*)d_in[7];
    const float* W_ho0  = (const float*)d_in[8];
    const float* b_ho0  = (const float*)d_in[9];
    const float* W_ih1  = (const float*)d_in[10];
    const float* b_ih1  = (const float*)d_in[11];
    const float* W_hh1  = (const float*)d_in[12];
    const float* b_hh1  = (const float*)d_in[13];
    const float* W_ho1  = (const float*)d_in[14];
    const float* b_ho1  = (const float*)d_in[15];

    char* ws = (char*)d_ws;
    float*          kern = (float*)(ws);
    unsigned*       bar  = (unsigned*)(ws + OFF_BAR);
    unsigned short* xe   = (unsigned short*)(ws + OFF_XE);
    unsigned short* xt   = (unsigned short*)(ws + OFF_XT);
    float*          pre  = (float*)(ws + OFF_PRE);
    unsigned short* hs   = (unsigned short*)(ws + OFF_HS);
    float*          out  = (float*)d_out;

    const dim3 gGemm(Hn / 64, TBn / 128);           // (8, 1024)
    const dim3 gFilt(BHn / 256, Tn);                // (256, 1024)

    // ---- shared coefficients + embedding --------------------------------
    pi_kernel<<<1, Hn, 0, stream>>>(b_d, kern);
    embed_kernel<<<TBn, 128, 0, stream>>>(inputs, W_emb, b_emb, xe);

    // ---- layer 0 --------------------------------------------------------
    filter_kernel<<<gFilt, 256, 0, stream>>>(xe, kern, xt);
    gemm_kernel<false><<<gGemm, 256, 0, stream>>>(xt, W_ih0, b_ih0, b_hh0, pre, Hn);
    hipMemsetAsync(bar, 0, sizeof(unsigned), stream);
    recur_kernel<<<NWG_REC, 512, 0, stream>>>(pre, W_hh0, hs, bar);
    // y0 (layer-1 input) overwrites xe
    gemm_kernel<true><<<gGemm, 256, 0, stream>>>(hs + BHn, W_ho0, b_ho0, nullptr, xe, Hn);

    // ---- layer 1 --------------------------------------------------------
    filter_kernel<<<gFilt, 256, 0, stream>>>(xe, kern, xt);
    gemm_kernel<false><<<gGemm, 256, 0, stream>>>(xt, W_ih1, b_ih1, b_hh1, pre, Hn);
    hipMemsetAsync(bar, 0, sizeof(unsigned), stream);
    recur_kernel<<<NWG_REC, 512, 0, stream>>>(pre, W_hh1, hs, bar);

    // ---- final step readout --------------------------------------------
    final_kernel<<<(Bn * OUTn + 255) / 256, 256, 0, stream>>>(
        hs + (size_t)Tn * BHn, W_ho1, b_ho1, out);
}
